// FMoETransformerMLP_1958505087363
// MI455X (gfx1250) — matrix operand order repaired and checked by reading, not compiled, not run on hardware
//
#include <hip/hip_runtime.h>
#include <hip/hip_bf16.h>
#include <math.h>

#define SEQ     2048
#define BATCH   4
#define NTOK    (SEQ*BATCH)          // 8192 tokens
#define DMODEL  1024
#define DHID    4096
#define NEXP    8
#define TOPK    2
#define NSLOT   (NTOK*TOPK)          // 16384 (token, k) slots
#define MAXTILES (NSLOT/16 + NEXP)   // 1032 worst case 16-row tiles
#define PADSLOT (MAXTILES*16)        // 16512 padded slot rows

typedef __attribute__((ext_vector_type(16))) __bf16 v16bf;
typedef __attribute__((ext_vector_type(8)))  __bf16 v8bf;
typedef __attribute__((ext_vector_type(4)))  __bf16 v4bf;
typedef __attribute__((ext_vector_type(8)))  float  v8f;

// ---------------------------------------------------------------- utilities

__global__ void k_cast_bf16(const float* __restrict__ src,
                            __bf16* __restrict__ dst, int n4) {
    int i = blockIdx.x * blockDim.x + threadIdx.x;
    if (i < n4) {
        float4 f = ((const float4*)src)[i];
        v4bf o;
        o[0] = (__bf16)f.x; o[1] = (__bf16)f.y;
        o[2] = (__bf16)f.z; o[3] = (__bf16)f.w;
        ((v4bf*)dst)[i] = o;
    }
}

__global__ void k_init(int* __restrict__ counts, int* __restrict__ perm) {
    int i = blockIdx.x * blockDim.x + threadIdx.x;
    if (i < NEXP)    counts[i] = 0;
    if (i < PADSLOT) perm[i]   = 0;   // dummy (padded) slots read token 0
}

// ------------------------------------------------------------- gate (top-2)

__global__ __launch_bounds__(128)
void k_gate(const float* __restrict__ x, const float* __restrict__ gw,
            const float* __restrict__ gb, int* __restrict__ counts,
            int* __restrict__ eidx, float* __restrict__ escore,
            int* __restrict__ erank) {
    __shared__ float gws[NEXP * DMODEL];             // 32 KB
    for (int i = threadIdx.x; i < NEXP * DMODEL; i += blockDim.x)
        gws[i] = gw[i];
    __syncthreads();

    int wave = threadIdx.x >> 5, lane = threadIdx.x & 31;
    int token = blockIdx.x * 4 + wave;               // 4 waves per block

    float acc[NEXP];
#pragma unroll
    for (int e = 0; e < NEXP; e++) acc[e] = 0.f;
    for (int d = lane; d < DMODEL; d += 32) {
        float xv = x[(size_t)token * DMODEL + d];
#pragma unroll
        for (int e = 0; e < NEXP; e++) acc[e] += xv * gws[e * DMODEL + d];
    }
#pragma unroll
    for (int e = 0; e < NEXP; e++)
        for (int off = 16; off > 0; off >>= 1)
            acc[e] += __shfl_xor(acc[e], off);       // wave32 reduction

    if (lane == 0) {
        float lg[NEXP];
#pragma unroll
        for (int e = 0; e < NEXP; e++) lg[e] = acc[e] + gb[e];
        int e0 = 0;
        for (int e = 1; e < NEXP; e++) if (lg[e] > lg[e0]) e0 = e;   // first max
        int e1 = -1;
        for (int e = 0; e < NEXP; e++)
            if (e != e0 && (e1 < 0 || lg[e] > lg[e1])) e1 = e;
        float t  = expf(lg[e1] - lg[e0]);            // softmax over top-2
        float p0 = 1.f / (1.f + t), p1 = t / (1.f + t);
        eidx[token * 2]     = e0;  eidx[token * 2 + 1]   = e1;
        escore[token * 2]   = p0;  escore[token * 2 + 1] = p1;
        erank[token * 2]     = atomicAdd(&counts[e0], 1);
        erank[token * 2 + 1] = atomicAdd(&counts[e1], 1);
    }
}

// ------------------------------------------- bucket scan / tile map / scatter

__global__ void k_scan(const int* __restrict__ counts, int* __restrict__ poff,
                       int* __restrict__ tile_expert, int* __restrict__ tile_base,
                       int* __restrict__ total_tiles) {
    if (blockIdx.x == 0 && threadIdx.x == 0) {
        int pad = 0, t = 0;
        for (int e = 0; e < NEXP; e++) {
            poff[e] = pad;
            int nt = (counts[e] + 15) >> 4;
            for (int i = 0; i < nt; i++) {
                tile_expert[t] = e;
                tile_base[t]   = pad + i * 16;
                t++;
            }
            pad += nt * 16;
        }
        *total_tiles = t;
    }
}

__global__ void k_scatter(const int* __restrict__ eidx, const int* __restrict__ erank,
                          const int* __restrict__ poff, int* __restrict__ perm,
                          int* __restrict__ inv_pos) {
    int s = blockIdx.x * blockDim.x + threadIdx.x;
    if (s < NSLOT) {
        int pos = poff[eidx[s]] + erank[s];
        perm[pos]  = s >> 1;                          // slot -> token row
        inv_pos[s] = pos;                             // (token,k) -> slot row
    }
}

// --------------------------------------------------------- fused expert FFN

__device__ inline v16bf load_a_frag(const __bf16* base, int stride, int row, int k0) {
    // A fragment in this code's contiguous k order (B loads match): half-wave h of lane row m holds K = k0..k0+15
    const v8bf lo = *(const v8bf*)(base + row * stride + k0);
    const v8bf hi = *(const v8bf*)(base + row * stride + k0 + 8);
    v16bf a;
#pragma unroll
    for (int i = 0; i < 8; i++) { a[i] = lo[i]; a[i + 8] = hi[i]; }
    return a;
}

// branchless exact-GELU: 0.5x(1+erf(x/sqrt2)), A&S 7.1.26 erf (|err|<1.5e-7)
__device__ inline float gelu_exact(float v) {
    float z  = v * 0.70710678118654752440f;
    float az = fabsf(z);
    float t  = 1.0f / fmaf(0.3275911f, az, 1.0f);
    float p  = fmaf(1.061405429f, t, -1.453152027f);
    p = fmaf(p, t,  1.421413741f);
    p = fmaf(p, t, -0.230389615f);
    p = fmaf(p, t,  0.254829592f);
    p = p * t;
    float erf_abs = fmaf(-p, __expf(-z * z), 1.0f);
    float erfv    = copysignf(erf_abs, z);
    return 0.5f * v * (1.0f + erfv);
}

__global__ __launch_bounds__(256)
void k_ffn(const __bf16* __restrict__ xb, const __bf16* __restrict__ w1b,
           const __bf16* __restrict__ w2b, const int* __restrict__ perm,
           const int* __restrict__ tile_expert, const int* __restrict__ tile_base,
           const int* __restrict__ total_tiles, float* __restrict__ y) {
    extern __shared__ __bf16 smem[];
    __bf16* Xs = smem;                    // 16 x 1024 bf16 = 32 KB
    __bf16* Hs = smem + 16 * DMODEL;      // 16 x 4096 bf16 = 128 KB

    int t = blockIdx.x;
    if (t >= *total_tiles) return;
    int e = tile_expert[t], base = tile_base[t];

    // gather 16 token rows into LDS (v8bf chunks)
    for (int i = threadIdx.x; i < 16 * (DMODEL / 8); i += 256) {
        int r = i / (DMODEL / 8), c = i % (DMODEL / 8);
        int tok = perm[base + r];
        *(v8bf*)(Xs + r * DMODEL + c * 8) =
            *(const v8bf*)(xb + (size_t)tok * DMODEL + c * 8);
    }
    __syncthreads();

    int wave = threadIdx.x >> 5, lane = threadIdx.x & 31;
    int half = lane >> 4, ln = lane & 15;

    // phase 1: H = gelu(X @ W1_e^T). 256 col-tiles; each wave owns groups of
    // 4 consecutive tiles so one A-fragment feeds 4 WMMAs.
    const __bf16* W1e = w1b + (size_t)e * DHID * DMODEL;
    for (int g = 0; g < 8; g++) {
        int h0 = (wave + 8 * g) * 64;                 // 4 tiles: h0 .. h0+63
        v8f acc0 = {}, acc1 = {}, acc2 = {}, acc3 = {};
        const __bf16* b0 = W1e + (size_t)(h0 +  0 + ln) * DMODEL + half * 16;
        const __bf16* b1 = W1e + (size_t)(h0 + 16 + ln) * DMODEL + half * 16;
        const __bf16* b2 = W1e + (size_t)(h0 + 32 + ln) * DMODEL + half * 16;
        const __bf16* b3 = W1e + (size_t)(h0 + 48 + ln) * DMODEL + half * 16;
#pragma unroll 4
        for (int kk = 0; kk < DMODEL / 32; kk++) {
            if ((kk & 3) == 0) {
                __builtin_prefetch(b0 + kk * 32 + 512, 0, 3);
                __builtin_prefetch(b1 + kk * 32 + 512, 0, 3);
                __builtin_prefetch(b2 + kk * 32 + 512, 0, 3);
                __builtin_prefetch(b3 + kk * 32 + 512, 0, 3);
            }
            v16bf a = load_a_frag(Xs, DMODEL, ln, kk * 32 + half * 16);
            acc0 = __builtin_amdgcn_wmma_f32_16x16x32_bf16(
                false, a, false, *(const v16bf*)(b0 + kk * 32), (short)0, acc0, false, false);
            acc1 = __builtin_amdgcn_wmma_f32_16x16x32_bf16(
                false, a, false, *(const v16bf*)(b1 + kk * 32), (short)0, acc1, false, false);
            acc2 = __builtin_amdgcn_wmma_f32_16x16x32_bf16(
                false, a, false, *(const v16bf*)(b2 + kk * 32), (short)0, acc2, false, false);
            acc3 = __builtin_amdgcn_wmma_f32_16x16x32_bf16(
                false, a, false, *(const v16bf*)(b3 + kk * 32), (short)0, acc3, false, false);
        }
#pragma unroll
        for (int j = 0; j < 8; j++) {                 // D row j+half*8, col +ln
            int r = (j + half * 8) * DHID + ln;
            Hs[r + h0 +  0] = (__bf16)gelu_exact(acc0[j]);
            Hs[r + h0 + 16] = (__bf16)gelu_exact(acc1[j]);
            Hs[r + h0 + 32] = (__bf16)gelu_exact(acc2[j]);
            Hs[r + h0 + 48] = (__bf16)gelu_exact(acc3[j]);
        }
    }
    __syncthreads();

    // phase 2: Y = H @ W2_e^T. 64 col-tiles; 2 groups of 4 per wave, K = 4096.
    const __bf16* W2e = w2b + (size_t)e * DMODEL * DHID;
    for (int g = 0; g < 2; g++) {
        int d0 = (wave + 8 * g) * 64;
        v8f acc0 = {}, acc1 = {}, acc2 = {}, acc3 = {};
        const __bf16* b0 = W2e + (size_t)(d0 +  0 + ln) * DHID + half * 16;
        const __bf16* b1 = W2e + (size_t)(d0 + 16 + ln) * DHID + half * 16;
        const __bf16* b2 = W2e + (size_t)(d0 + 32 + ln) * DHID + half * 16;
        const __bf16* b3 = W2e + (size_t)(d0 + 48 + ln) * DHID + half * 16;
#pragma unroll 4
        for (int kk = 0; kk < DHID / 32; kk++) {
            if ((kk & 3) == 0) {
                __builtin_prefetch(b0 + kk * 32 + 512, 0, 3);
                __builtin_prefetch(b1 + kk * 32 + 512, 0, 3);
                __builtin_prefetch(b2 + kk * 32 + 512, 0, 3);
                __builtin_prefetch(b3 + kk * 32 + 512, 0, 3);
            }
            v16bf a = load_a_frag(Hs, DHID, ln, kk * 32 + half * 16);
            acc0 = __builtin_amdgcn_wmma_f32_16x16x32_bf16(
                false, a, false, *(const v16bf*)(b0 + kk * 32), (short)0, acc0, false, false);
            acc1 = __builtin_amdgcn_wmma_f32_16x16x32_bf16(
                false, a, false, *(const v16bf*)(b1 + kk * 32), (short)0, acc1, false, false);
            acc2 = __builtin_amdgcn_wmma_f32_16x16x32_bf16(
                false, a, false, *(const v16bf*)(b2 + kk * 32), (short)0, acc2, false, false);
            acc3 = __builtin_amdgcn_wmma_f32_16x16x32_bf16(
                false, a, false, *(const v16bf*)(b3 + kk * 32), (short)0, acc3, false, false);
        }
#pragma unroll
        for (int j = 0; j < 8; j++) {
            size_t r = (size_t)(base + j + half * 8) * DMODEL + ln;
            y[r + d0 +  0] = acc0[j];
            y[r + d0 + 16] = acc1[j];
            y[r + d0 + 32] = acc2[j];
            y[r + d0 + 48] = acc3[j];
        }
    }
}

// ------------------------------------------ residual + combine + layer norm

__global__ __launch_bounds__(256)
void k_final(const float* __restrict__ inp, const float* __restrict__ y,
             const float* __restrict__ escore, const int* __restrict__ inv_pos,
             const float* __restrict__ gamma, const float* __restrict__ beta,
             float* __restrict__ out) {
    int n = blockIdx.x, tid = threadIdx.x;
    int p0 = inv_pos[n * 2], p1 = inv_pos[n * 2 + 1];
    float s0 = escore[n * 2], s1 = escore[n * 2 + 1];

    float v[4], sum = 0.f, sq = 0.f;
#pragma unroll
    for (int i = 0; i < 4; i++) {
        int d = tid + i * 256;
        float val = inp[(size_t)n * DMODEL + d]
                  + s0 * y[(size_t)p0 * DMODEL + d]
                  + s1 * y[(size_t)p1 * DMODEL + d];
        v[i] = val; sum += val; sq += val * val;
    }
    for (int off = 16; off > 0; off >>= 1) {
        sum += __shfl_xor(sum, off);
        sq  += __shfl_xor(sq,  off);
    }
    __shared__ float ssum[8], ssq[8];
    if ((tid & 31) == 0) { ssum[tid >> 5] = sum; ssq[tid >> 5] = sq; }
    __syncthreads();
    float ts = 0.f, tq = 0.f;
#pragma unroll
    for (int w = 0; w < 8; w++) { ts += ssum[w]; tq += ssq[w]; }
    float mean = ts * (1.f / DMODEL);
    float var  = tq * (1.f / DMODEL) - mean * mean;
    float rstd = rsqrtf(var + 1e-5f);
#pragma unroll
    for (int i = 0; i < 4; i++) {
        int d = tid + i * 256;
        out[(size_t)n * DMODEL + d] = (v[i] - mean) * rstd * gamma[d] + beta[d];
    }
}

__global__ void k_bias_tail(const float* __restrict__ mb, float* __restrict__ out) {
    int d = blockIdx.x * blockDim.x + threadIdx.x;
    if (d < DMODEL) out[(size_t)NTOK * DMODEL + d] = mb[d];
}

// ------------------------------------------------------------------- launch

extern "C" void kernel_launch(void* const* d_in, const int* in_sizes, int n_in,
                              void* d_out, int out_size, void* d_ws, size_t ws_size,
                              hipStream_t stream) {
    const float* inp      = (const float*)d_in[0];
    const float* gate_w   = (const float*)d_in[1];
    const float* gate_b   = (const float*)d_in[2];
    const float* w1       = (const float*)d_in[3];
    const float* w2       = (const float*)d_in[4];
    const float* ln_gamma = (const float*)d_in[5];
    const float* ln_beta  = (const float*)d_in[6];
    const float* moe_bias = (const float*)d_in[7];
    float* out = (float*)d_out;

    char* ws = (char*)d_ws;
    size_t off = 0;
    auto carve = [&](size_t bytes) -> void* {
        void* p = (void*)(ws + off);
        off = (off + bytes + 255) & ~(size_t)255;
        return p;
    };
    __bf16* xb  = (__bf16*)carve((size_t)NTOK * DMODEL * 2);
    __bf16* w1b = (__bf16*)carve((size_t)NEXP * DHID * DMODEL * 2);
    __bf16* w2b = (__bf16*)carve((size_t)NEXP * DMODEL * DHID * 2);
    float*  y   = (float*) carve((size_t)PADSLOT * DMODEL * 4);
    int*   counts      = (int*)  carve(NEXP * 4);
    int*   poff        = (int*)  carve(NEXP * 4);
    int*   total_tiles = (int*)  carve(4);
    int*   tile_expert = (int*)  carve(MAXTILES * 4);
    int*   tile_base   = (int*)  carve(MAXTILES * 4);
    int*   eidx        = (int*)  carve(NSLOT * 4);
    float* escore      = (float*)carve(NSLOT * 4);
    int*   erank       = (int*)  carve(NSLOT * 4);
    int*   perm        = (int*)  carve(PADSLOT * 4);
    int*   inv_pos     = (int*)  carve(NSLOT * 4);

    int n4x  = NTOK * DMODEL / 4;
    int n4w  = NEXP * DHID * DMODEL / 4;
    k_cast_bf16<<<(n4x + 255) / 256, 256, 0, stream>>>(inp, xb, n4x);
    k_cast_bf16<<<(n4w + 255) / 256, 256, 0, stream>>>(w1, w1b, n4w);
    k_cast_bf16<<<(n4w + 255) / 256, 256, 0, stream>>>(w2, w2b, n4w);

    k_init<<<(PADSLOT + 255) / 256, 256, 0, stream>>>(counts, perm);
    k_gate<<<NTOK / 4, 128, 0, stream>>>(inp, gate_w, gate_b,
                                         counts, eidx, escore, erank);
    k_scan<<<1, 1, 0, stream>>>(counts, poff, tile_expert, tile_base, total_tiles);
    k_scatter<<<(NSLOT + 255) / 256, 256, 0, stream>>>(eidx, erank, poff,
                                                       perm, inv_pos);

    size_t lds_bytes = (size_t)(16 * DMODEL + 16 * DHID) * sizeof(__bf16); // 160 KB
    k_ffn<<<MAXTILES, 256, lds_bytes, stream>>>(xb, w1b, w2b, perm,
                                                tile_expert, tile_base,
                                                total_tiles, y);

    k_final<<<NTOK, 256, 0, stream>>>(inp, y, escore, inv_pos,
                                      ln_gamma, ln_beta, out);
    k_bias_tail<<<(DMODEL + 255) / 256, 256, 0, stream>>>(moe_bias, out);
}
